// LogisticRegressionModel_40226663694549
// MI455X (gfx1250) — matrix operand order repaired and checked by reading, not compiled, not run on hardware
//
#include <hip/hip_runtime.h>

typedef __attribute__((ext_vector_type(16))) __bf16 v16bf;
typedef __attribute__((ext_vector_type(8)))  float  v8f;
typedef int v4i_raw __attribute__((__vector_size__(16)));

#define GEMM_B 16384
#define GEMM_F 4096
#define GEMM_C 8192
#define TILE_K 32
#define NKT    (GEMM_F / TILE_K)      // 128 K-tiles
#define NMT    (GEMM_B / 16)          // 1024 M-subtiles
#define NNT    (GEMM_C / 16)          // 512  N-subtiles

#if defined(__HIP_DEVICE_COMPILE__) && __has_builtin(__builtin_amdgcn_global_load_async_to_lds_b128)
#define HAVE_ASYNC_LDS 1
#else
#define HAVE_ASYNC_LDS 0
#endif

#define AS1P(p) ((__attribute__((address_space(1))) v4i_raw*)(void*)(p))
#define AS3P(p) ((__attribute__((address_space(3))) v4i_raw*)(void*)(p))

__device__ __forceinline__ void wait_async_copies() {
#if HAVE_ASYNC_LDS
#if __has_builtin(__builtin_amdgcn_s_wait_asynccnt)
    __builtin_amdgcn_s_wait_asynccnt(0);
#else
    asm volatile("s_wait_asynccnt 0x0" ::: "memory");
#endif
#endif
}

__device__ __forceinline__ unsigned short f32_to_bf16_rne(float f) {
    unsigned u = __float_as_uint(f);
    unsigned r = u + 0x7FFFu + ((u >> 16) & 1u);   // round-to-nearest-even
    return (unsigned short)(r >> 16);
}
__device__ __forceinline__ float bf16_bits_to_f32(unsigned short h) {
    return __uint_as_float(((unsigned)h) << 16);
}
__device__ __forceinline__ void split_bf16(float f, unsigned short& hi, unsigned short& lo) {
    hi = f32_to_bf16_rne(f);
    lo = f32_to_bf16_rne(f - bf16_bits_to_f32(hi));
}

// ---------------------------------------------------------------------------
// Pass 1a: A[16384,4096] fp32 -> Ah/Al bf16, fragment-swizzled [kt][mt][lane][16]
// A-frag layout (16-bit 16x32): lanes 0-15 hold K{0..7,16..23}, lanes 16-31 K{8..15,24..31}
// ---------------------------------------------------------------------------
__global__ __launch_bounds__(256)
void convert_swizzle_A(const float* __restrict__ A,
                       unsigned short* __restrict__ Ah,
                       unsigned short* __restrict__ Al) {
    __shared__ uint4 sh[512], sl[512];             // 8 KiB each
    const int tid = threadIdx.x;
    const int bm  = blockIdx.x * 128;
    const int kt  = blockIdx.y;
    const int k0  = kt * TILE_K;
#pragma unroll
    for (int i = 0; i < 4; ++i) {
        const int slot = tid + 256 * i;            // 1024 float4 slots (128 x 32)
        const int m  = slot >> 3;
        const int k4 = (slot & 7) << 2;
        const float4 av = *(const float4*)(A + (size_t)(bm + m) * GEMM_F + (k0 + k4));
        ushort4 hi, lo;
        split_bf16(av.x, hi.x, lo.x);
        split_bf16(av.y, hi.y, lo.y);
        split_bf16(av.z, hi.z, lo.z);
        split_bf16(av.w, hi.w, lo.w);
        const int tileM = m >> 4;
        const int la    = (m & 15) + (((k4 >> 3) & 1) << 4);
        const int h0    = (k4 & 7) + (((k4 >> 4) & 1) << 3);
        const int off   = tileM * 512 + la * 16 + h0;  // ushort units
        *(ushort4*)((unsigned short*)sh + off) = hi;
        *(ushort4*)((unsigned short*)sl + off) = lo;
    }
    __syncthreads();
    const size_t dst = ((size_t)kt * NMT + (bm >> 4)) * 512;   // ushorts, 1KiB-aligned
    uint4* gh = (uint4*)(Ah + dst);
    uint4* gl = (uint4*)(Al + dst);
    gh[tid] = sh[tid]; gh[tid + 256] = sh[tid + 256];
    gl[tid] = sl[tid]; gl[tid + 256] = sl[tid + 256];
}

// ---------------------------------------------------------------------------
// Pass 1b: W[4096,8192] fp32 -> Wh/Wl bf16, fragment-swizzled [kt][nt][lane][16]
// B-frag layout (16-bit 32x16): lanes 0-15 hold K{0..7,16..23}, lanes 16-31 K{8..15,24..31}
// ---------------------------------------------------------------------------
__global__ __launch_bounds__(256)
void convert_swizzle_W(const float* __restrict__ W,
                       unsigned short* __restrict__ Wh,
                       unsigned short* __restrict__ Wl) {
    __shared__ uint4 sh[512], sl[512];
    const int tid = threadIdx.x;
    const int bn  = blockIdx.x * 128;
    const int kt  = blockIdx.y;
    const int k0  = kt * TILE_K;
#pragma unroll
    for (int i = 0; i < 4; ++i) {
        const int slot = tid + 256 * i;            // 1024 float4 slots (32 x 128)
        const int kk = slot >> 5;
        const int n4 = (slot & 31) << 2;
        const float4 bv = *(const float4*)(W + (size_t)(k0 + kk) * GEMM_C + (bn + n4));
        const int tileN = n4 >> 4;
        const int lb0   = (n4 & 15) + (((kk >> 3) & 1) << 4);
        const int h     = (kk & 7) + (((kk >> 4) & 1) << 3);
        const int base  = tileN * 512 + lb0 * 16 + h;
        unsigned short hx, lx;
        unsigned short* psh = (unsigned short*)sh;
        unsigned short* psl = (unsigned short*)sl;
        split_bf16(bv.x, hx, lx); psh[base]      = hx; psl[base]      = lx;
        split_bf16(bv.y, hx, lx); psh[base + 16] = hx; psl[base + 16] = lx;
        split_bf16(bv.z, hx, lx); psh[base + 32] = hx; psl[base + 32] = lx;
        split_bf16(bv.w, hx, lx); psh[base + 48] = hx; psl[base + 48] = lx;
    }
    __syncthreads();
    const size_t dst = ((size_t)kt * NNT + (bn >> 4)) * 512;
    uint4* gh = (uint4*)(Wh + dst);
    uint4* gl = (uint4*)(Wl + dst);
    gh[tid] = sh[tid]; gh[tid + 256] = sh[tid + 256];
    gl[tid] = sl[tid]; gl[tid + 256] = sl[tid + 256];
}

// ---------------------------------------------------------------------------
// Pass 2: GEMM on pre-swizzled bf16 hi/lo. 128 threads = 4 waves, each wave
// owns a 64x64 output tile (4x4 WMMA subtiles): 48 WMMA per 32 ds_load_b128.
// Double-buffered 64 KiB LDS fed by gfx1250 async global->LDS DMA.
// ---------------------------------------------------------------------------
__global__ __launch_bounds__(128, 1)
void gemm_staged_bf16x3(const unsigned short* __restrict__ Ah,
                        const unsigned short* __restrict__ Al,
                        const unsigned short* __restrict__ Wh,
                        const unsigned short* __restrict__ Wl,
                        const float* __restrict__ bias,
                        float* __restrict__ out) {
    __shared__ uint4 lsAh[2][512], lsAl[2][512], lsBh[2][512], lsBl[2][512];  // 64 KiB

    const int tid  = threadIdx.x;
    const int lane = tid & 31;
    const int wave = tid >> 5;           // 0..3
    const int wm   = wave >> 1;          // 0..1 : 64-row band
    const int wn   = wave & 1;           // 0..1 : 64-col band
    const int bm   = blockIdx.y * 128;
    const int bn   = blockIdx.x * 128;
    const int mt0  = bm >> 4;
    const int nt0  = bn >> 4;

    v8f acc[4][4] = {};                  // 4 M-subtiles x 4 N-subtiles of 16x16 f32

    auto frag = [&](const uint4* buf, int t) -> v16bf {
        return *((const v16bf*)buf + t * 32 + lane);
    };
    auto compute = [&](int buf) {
        v16bf aH[4], aL[4];
#pragma unroll
        for (int im = 0; im < 4; ++im) {
            const int t = wm * 4 + im;
            aH[im] = frag(lsAh[buf], t);
            aL[im] = frag(lsAl[buf], t);
        }
#pragma unroll
        for (int jn = 0; jn < 4; ++jn) {
            v16bf bh = frag(lsBh[buf], wn * 4 + jn);
            v16bf bl = frag(lsBl[buf], wn * 4 + jn);
#pragma unroll
            for (int im = 0; im < 4; ++im)
                acc[im][jn] = __builtin_amdgcn_wmma_f32_16x16x32_bf16(
                    false, aH[im], false, bh, (short)0, acc[im][jn], false, false);
#pragma unroll
            for (int im = 0; im < 4; ++im)
                acc[im][jn] = __builtin_amdgcn_wmma_f32_16x16x32_bf16(
                    false, aH[im], false, bl, (short)0, acc[im][jn], false, false);
#pragma unroll
            for (int im = 0; im < 4; ++im)
                acc[im][jn] = __builtin_amdgcn_wmma_f32_16x16x32_bf16(
                    false, aL[im], false, bh, (short)0, acc[im][jn], false, false);
        }
    };

#if HAVE_ASYNC_LDS
    // -------- gfx1250 async global->LDS DMA staging (no staging VGPRs) --------
    // 128 threads x 16B x 4 sweeps covers each 8 KiB tile; the literal imm
    // offset moves both the global and LDS side per the ISA addressing rules.
    auto issue = [&](int kt, int buf) {
        const uint4* pa = (const uint4*)(Ah + ((size_t)kt * NMT + mt0) * 512) + tid;
        const uint4* pl = (const uint4*)(Al + ((size_t)kt * NMT + mt0) * 512) + tid;
        const uint4* pb = (const uint4*)(Wh + ((size_t)kt * NNT + nt0) * 512) + tid;
        const uint4* pw = (const uint4*)(Wl + ((size_t)kt * NNT + nt0) * 512) + tid;
        __builtin_amdgcn_global_load_async_to_lds_b128(AS1P(pa), AS3P(&lsAh[buf][tid]), 0,    0);
        __builtin_amdgcn_global_load_async_to_lds_b128(AS1P(pa), AS3P(&lsAh[buf][tid]), 2048, 0);
        __builtin_amdgcn_global_load_async_to_lds_b128(AS1P(pa), AS3P(&lsAh[buf][tid]), 4096, 0);
        __builtin_amdgcn_global_load_async_to_lds_b128(AS1P(pa), AS3P(&lsAh[buf][tid]), 6144, 0);
        __builtin_amdgcn_global_load_async_to_lds_b128(AS1P(pl), AS3P(&lsAl[buf][tid]), 0,    0);
        __builtin_amdgcn_global_load_async_to_lds_b128(AS1P(pl), AS3P(&lsAl[buf][tid]), 2048, 0);
        __builtin_amdgcn_global_load_async_to_lds_b128(AS1P(pl), AS3P(&lsAl[buf][tid]), 4096, 0);
        __builtin_amdgcn_global_load_async_to_lds_b128(AS1P(pl), AS3P(&lsAl[buf][tid]), 6144, 0);
        __builtin_amdgcn_global_load_async_to_lds_b128(AS1P(pb), AS3P(&lsBh[buf][tid]), 0,    0);
        __builtin_amdgcn_global_load_async_to_lds_b128(AS1P(pb), AS3P(&lsBh[buf][tid]), 2048, 0);
        __builtin_amdgcn_global_load_async_to_lds_b128(AS1P(pb), AS3P(&lsBh[buf][tid]), 4096, 0);
        __builtin_amdgcn_global_load_async_to_lds_b128(AS1P(pb), AS3P(&lsBh[buf][tid]), 6144, 0);
        __builtin_amdgcn_global_load_async_to_lds_b128(AS1P(pw), AS3P(&lsBl[buf][tid]), 0,    0);
        __builtin_amdgcn_global_load_async_to_lds_b128(AS1P(pw), AS3P(&lsBl[buf][tid]), 2048, 0);
        __builtin_amdgcn_global_load_async_to_lds_b128(AS1P(pw), AS3P(&lsBl[buf][tid]), 4096, 0);
        __builtin_amdgcn_global_load_async_to_lds_b128(AS1P(pw), AS3P(&lsBl[buf][tid]), 6144, 0);
    };

    issue(0, 0);
    wait_async_copies();
    __syncthreads();
    for (int kt = 0; kt < NKT; kt += 2) {
        issue(kt + 1, 1);                 // NKT even: kt+1 always valid
        compute(0);
        wait_async_copies();
        __syncthreads();

        if (kt + 2 < NKT) issue(kt + 2, 0);
        compute(1);
        wait_async_copies();
        __syncthreads();
    }
#else
    // -------- fallback: register-staged copy --------
    uint4 r[4][4];
    auto glo = [&](int kt) {
        const uint4* pa = (const uint4*)(Ah + ((size_t)kt * NMT + mt0) * 512);
        const uint4* pl = (const uint4*)(Al + ((size_t)kt * NMT + mt0) * 512);
        const uint4* pb = (const uint4*)(Wh + ((size_t)kt * NNT + nt0) * 512);
        const uint4* pw = (const uint4*)(Wl + ((size_t)kt * NNT + nt0) * 512);
#pragma unroll
        for (int o = 0; o < 4; ++o) {
            r[0][o] = pa[tid + o * 128];
            r[1][o] = pl[tid + o * 128];
            r[2][o] = pb[tid + o * 128];
            r[3][o] = pw[tid + o * 128];
        }
    };
    auto sto = [&](int buf) {
#pragma unroll
        for (int o = 0; o < 4; ++o) {
            lsAh[buf][tid + o * 128] = r[0][o];
            lsAl[buf][tid + o * 128] = r[1][o];
            lsBh[buf][tid + o * 128] = r[2][o];
            lsBl[buf][tid + o * 128] = r[3][o];
        }
    };
    glo(0); sto(0);
    __syncthreads();
    for (int kt = 0; kt < NKT; kt += 2) {
        glo(kt + 1);
        compute(0);
        sto(1);
        __syncthreads();

        if (kt + 2 < NKT) glo(kt + 2);
        compute(1);
        if (kt + 2 < NKT) sto(0);
        __syncthreads();
    }
#endif

    // Epilogue: bias add + store (C/D layout: lane -> N, vgpr index -> M)
    const int nn = lane & 15;
    const int hf = lane >> 4;
#pragma unroll
    for (int jn = 0; jn < 4; ++jn) {
        const int n   = bn + wn * 64 + jn * 16 + nn;
        const float bv = bias[n];
#pragma unroll
        for (int im = 0; im < 4; ++im) {
            const int m0 = bm + wm * 64 + im * 16 + hf * 8;
#pragma unroll
            for (int r8 = 0; r8 < 8; ++r8) {
                out[(size_t)(m0 + r8) * GEMM_C + n] = acc[im][jn][r8] + bv;
            }
        }
    }
}

// ---------------------------------------------------------------------------
// Fallback: fused single-pass kernel (in-loop conversion), used if ws too small.
// ---------------------------------------------------------------------------
__global__ __launch_bounds__(256, 1)
void logreg_gemm_bias_fused(const float* __restrict__ A,
                            const float* __restrict__ W,
                            const float* __restrict__ bias,
                            float* __restrict__ out) {
    __shared__ __align__(16) unsigned short lsAh[4096];
    __shared__ __align__(16) unsigned short lsAl[4096];
    __shared__ __align__(16) unsigned short lsBh[4096];
    __shared__ __align__(16) unsigned short lsBl[4096];

    const int tid  = threadIdx.x;
    const int lane = tid & 31;
    const int wave = tid >> 5;
    const int wm   = wave >> 1;
    const int wn   = wave & 1;
    const int bm   = blockIdx.y * 128;
    const int bn   = blockIdx.x * 128;

    v8f acc[2][4] = {};

    for (int k0 = 0; k0 < GEMM_F; k0 += TILE_K) {
#pragma unroll
        for (int i = 0; i < 4; ++i) {
            const int slot = tid + 256 * i;
            {
                const int m  = slot >> 3;
                const int k4 = (slot & 7) << 2;
                const float4 av = *(const float4*)(A + (size_t)(bm + m) * GEMM_F + (k0 + k4));
                ushort4 hi, lo;
                split_bf16(av.x, hi.x, lo.x);
                split_bf16(av.y, hi.y, lo.y);
                split_bf16(av.z, hi.z, lo.z);
                split_bf16(av.w, hi.w, lo.w);
                const int off = (m >> 4) * 512 + ((m & 15) + (((k4 >> 3) & 1) << 4)) * 16
                              + (k4 & 7) + (((k4 >> 4) & 1) << 3);
                *(ushort4*)(&lsAh[off]) = hi;
                *(ushort4*)(&lsAl[off]) = lo;
            }
            {
                const int kk = slot >> 5;
                const int n4 = (slot & 31) << 2;
                const float4 bv = *(const float4*)(W + (size_t)(k0 + kk) * GEMM_C + (bn + n4));
                const int base = (n4 >> 4) * 512 + ((n4 & 15) + (((kk >> 3) & 1) << 4)) * 16 + (kk & 7) + (((kk >> 4) & 1) << 3);
                unsigned short hx, lx;
                split_bf16(bv.x, hx, lx); lsBh[base]      = hx; lsBl[base]      = lx;
                split_bf16(bv.y, hx, lx); lsBh[base + 16] = hx; lsBl[base + 16] = lx;
                split_bf16(bv.z, hx, lx); lsBh[base + 32] = hx; lsBl[base + 32] = lx;
                split_bf16(bv.w, hx, lx); lsBh[base + 48] = hx; lsBl[base + 48] = lx;
            }
        }
        __syncthreads();

        v16bf aH[2], aL[2], bH[4], bL[4];
#pragma unroll
        for (int im = 0; im < 2; ++im) {
            const int t = wm * 2 + im;
            aH[im] = *((const v16bf*)lsAh + t * 32 + lane);
            aL[im] = *((const v16bf*)lsAl + t * 32 + lane);
        }
#pragma unroll
        for (int jn = 0; jn < 4; ++jn) {
            const int t = wn * 4 + jn;
            bH[jn] = *((const v16bf*)lsBh + t * 32 + lane);
            bL[jn] = *((const v16bf*)lsBl + t * 32 + lane);
        }
#pragma unroll
        for (int im = 0; im < 2; ++im) {
#pragma unroll
            for (int jn = 0; jn < 4; ++jn) {
                acc[im][jn] = __builtin_amdgcn_wmma_f32_16x16x32_bf16(
                    false, aH[im], false, bH[jn], (short)0, acc[im][jn], false, false);
                acc[im][jn] = __builtin_amdgcn_wmma_f32_16x16x32_bf16(
                    false, aH[im], false, bL[jn], (short)0, acc[im][jn], false, false);
                acc[im][jn] = __builtin_amdgcn_wmma_f32_16x16x32_bf16(
                    false, aL[im], false, bH[jn], (short)0, acc[im][jn], false, false);
            }
        }
        __syncthreads();
    }

    const int nn = lane & 15;
    const int hf = lane >> 4;
#pragma unroll
    for (int jn = 0; jn < 4; ++jn) {
        const int n   = bn + wn * 64 + jn * 16 + nn;
        const float bv = bias[n];
#pragma unroll
        for (int im = 0; im < 2; ++im) {
            const int m0 = bm + wm * 32 + im * 16 + hf * 8;
#pragma unroll
            for (int r8 = 0; r8 < 8; ++r8) {
                out[(size_t)(m0 + r8) * GEMM_C + n] = acc[im][jn][r8] + bv;
            }
        }
    }
}

extern "C" void kernel_launch(void* const* d_in, const int* in_sizes, int n_in,
                              void* d_out, int out_size, void* d_ws, size_t ws_size,
                              hipStream_t stream) {
    const float* inp  = (const float*)d_in[0];   // [16384, 4096]
    const float* wgt  = (const float*)d_in[1];   // [4096, 8192]
    const float* bias = (const float*)d_in[2];   // [8192]
    float* out = (float*)d_out;                  // [16384, 8192]

    const size_t NA = (size_t)GEMM_B * GEMM_F;   // 67,108,864
    const size_t NW = (size_t)GEMM_F * GEMM_C;   // 33,554,432
    const size_t need = (2 * NA + 2 * NW) * sizeof(unsigned short);  // 384 MiB

    dim3 ggrid(GEMM_C / 128, GEMM_B / 128);      // (64, 128)

    if (ws_size >= need) {
        unsigned short* Ah = (unsigned short*)d_ws;
        unsigned short* Al = Ah + NA;
        unsigned short* Wh = Al + NA;
        unsigned short* Wl = Wh + NW;
        convert_swizzle_A<<<dim3(GEMM_B / 128, NKT), 256, 0, stream>>>(inp, Ah, Al);
        convert_swizzle_W<<<dim3(GEMM_C / 128, NKT), 256, 0, stream>>>(wgt, Wh, Wl);
        gemm_staged_bf16x3<<<ggrid, 128, 0, stream>>>(Ah, Al, Wh, Wl, bias, out);
    } else {
        logreg_gemm_bias_fused<<<ggrid, 256, 0, stream>>>(inp, wgt, bias, out);
    }
}
